// MultiHeadSelfAttention_12335146074390
// MI455X (gfx1250) — matrix operand order repaired and checked by reading, not compiled, not run on hardware
//
#include <hip/hip_runtime.h>

// ---------------- problem constants ----------------
constexpr int Bb = 4;      // batch
constexpr int Tq = 2048;   // sequence length
constexpr int DM = 1024;   // d_model
constexpr int NH = 16;     // heads
constexpr int DH = 64;     // head dim
constexpr float NEG_BIG = -1e30f;

// ---------------- WMMA fragment types --------------
typedef __attribute__((ext_vector_type(16))) __bf16 v16bf;
typedef __attribute__((ext_vector_type(8)))  float  v8f;
typedef __attribute__((ext_vector_type(4))) unsigned v4u;
typedef __attribute__((ext_vector_type(8))) int      v8i;
typedef __attribute__((ext_vector_type(4))) int      v4i;

union Frag16 { v16bf v; unsigned u[8]; unsigned short h[16]; };
union FragC  { v8f   v; float f[8]; };

__device__ __forceinline__ unsigned short f2bf(float x) {
  unsigned u = __float_as_uint(x);
  u += 0x7fffu + ((u >> 16) & 1u);   // round-to-nearest-even
  return (unsigned short)(u >> 16);
}

__device__ __forceinline__ v8f wmma_bf16(const Frag16& a, const Frag16& b, v8f c) {
  return __builtin_amdgcn_wmma_f32_16x16x32_bf16(
      /*neg_a=*/false, a.v, /*neg_b=*/false, b.v,
      /*c_mod=*/(short)0, c, /*reuse_a=*/false, /*reuse_b=*/false);
}

// A-matrix (16x32, 16-bit) per-lane packed-K element offset for VGPR j.
__device__ __forceinline__ int a_koff(int j, int hA) {
  return 2 * j + 16 * hA;
}

// ---- Tensor Data Mover: 2D bf16 tile (global -> LDS), D# per ISA 08 ----
// tile_d0 elements contiguous per row, tile_d1 rows, row pitch stride0 elems.
__device__ __forceinline__ void tdm_load_2d(unsigned lds_off,
                                            const unsigned short* gptr,
                                            unsigned tile_d0, unsigned tile_d1,
                                            unsigned stride0) {
  unsigned long long ga = (unsigned long long)(const void*)gptr;
  v4u g0;
  g0.x = 1u;                                       // count=1 (valid), no gather
  g0.y = lds_off;                                  // lds_addr (bytes)
  g0.z = (unsigned)(ga & 0xffffffffu);             // global_addr[31:0]
  g0.w = (unsigned)((ga >> 32) & 0x01ffffffu)      // global_addr[56:32]
         | (2u << 30);                             // type=2 ("image")
  v8i g1;
  g1[0] = (int)(1u << 16);                         // data_size=1 -> 2 bytes
  g1[1] = (int)((tile_d0 & 0xffffu) << 16);        // tensor_dim0[15:0]  (=tile)
  g1[2] = (int)((tile_d0 >> 16) |
                ((tile_d1 & 0xffffu) << 16));      // tensor_dim0[31:16], dim1[15:0]
  g1[3] = (int)((tile_d1 >> 16) |
                ((tile_d0 & 0xffffu) << 16));      // tensor_dim1[31:16], tile_dim0
  g1[4] = (int)(tile_d1 & 0xffffu);                // tile_dim1 (tile_dim2=0)
  g1[5] = (int)stride0;                            // tensor_dim0_stride[31:0]
  g1[6] = 0;                                       // stride0[47:32], dim1_stride lo
  g1[7] = 0;                                       // dim1_stride hi
  v4i z4 = {0, 0, 0, 0};
  v8i z8 = {0, 0, 0, 0, 0, 0, 0, 0};
  // 6-arg toolchain form: (g0, g1, g2, g3, extra, cpol)
  __builtin_amdgcn_tensor_load_to_lds(g0, g1, z4, z4, z8, 0);
}

// ---------------- kernel 1: f32 -> bf16 cast ----------------
__global__ void cast_bf16(const float* __restrict__ src,
                          unsigned short* __restrict__ dst, int n) {
  int i = blockIdx.x * blockDim.x + threadIdx.x;
  if (i < n) dst[i] = f2bf(src[i]);
}

// ---------------- kernel 2: QKV projection GEMM -------------
// C[8192,3072] = X * Wqkv^T.  Q scattered [B][H][T][DH] (scaled 0.125),
// K scattered [B][H][T][DH],  V scattered TRANSPOSED [B][H][DH][T].
__global__ __launch_bounds__(256) void qkv_gemm(
    const unsigned short* __restrict__ xb,
    const unsigned short* __restrict__ wb,
    unsigned short* __restrict__ qkv) {
  const int lane = threadIdx.x & 31;
  const int wave = threadIdx.x >> 5;
  const int wm = wave >> 2, wn = wave & 3;
  const int row0 = blockIdx.x * 64 + wm * 32;
  const int col0 = blockIdx.y * 128 + wn * 32;
  const int rA = lane & 15, hA = lane >> 4;

  FragC acc[2][2];
#pragma unroll
  for (int i = 0; i < 2; ++i)
#pragma unroll
    for (int j = 0; j < 2; ++j)
#pragma unroll
      for (int r = 0; r < 8; ++r) acc[i][j].f[r] = 0.0f;

  for (int k = 0; k < DM; k += 32) {
    Frag16 a[2], b[2];
#pragma unroll
    for (int i = 0; i < 2; ++i) {
      const unsigned short* ap = xb + (size_t)(row0 + 16 * i + rA) * DM + k;
      __builtin_prefetch(ap + 32, 0, 1);
#pragma unroll
      for (int j = 0; j < 8; ++j)
        a[i].u[j] = *(const unsigned*)(ap + a_koff(j, hA));
    }
#pragma unroll
    for (int i = 0; i < 2; ++i) {
      const unsigned short* bp =
          wb + (size_t)(col0 + 16 * i + rA) * DM + k + 16 * hA;
      __builtin_prefetch(bp + 32, 0, 1);
#pragma unroll
      for (int j = 0; j < 8; ++j)
        b[i].u[j] = *(const unsigned*)(bp + 2 * j);
    }
#pragma unroll
    for (int i = 0; i < 2; ++i)
#pragma unroll
      for (int j = 0; j < 2; ++j)
        acc[i][j].v = wmma_bf16(a[i], b[j], acc[i][j].v);
  }

  const size_t plane = (size_t)Bb * NH * Tq * DH;
#pragma unroll
  for (int i = 0; i < 2; ++i)
#pragma unroll
    for (int j = 0; j < 2; ++j)
#pragma unroll
      for (int r = 0; r < 8; ++r) {
        int m = row0 + 16 * i + r + 8 * hA;
        int n = col0 + 16 * j + rA;
        int bidx = m >> 11, t = m & (Tq - 1);
        int which = n >> 10, dloc = n & (DM - 1);
        int h = dloc >> 6, dd = dloc & (DH - 1);
        float val = acc[i][j].f[r];
        size_t off;
        if (which == 0) {
          val *= 0.125f;                            // fold 1/sqrt(DH) into Q
          off = ((size_t)((bidx * NH + h) * Tq + t)) * DH + dd;
        } else if (which == 1) {
          off = plane + ((size_t)((bidx * NH + h) * Tq + t)) * DH + dd;
        } else {                                    // V transposed: [B][H][DH][T]
          off = 2 * plane + ((size_t)((bidx * NH + h) * DH + dd)) * Tq + t;
        }
        qkv[off] = f2bf(val);
      }
}

// ---------------- kernel 3: fused causal flash attention ----
// Block = one (b,h) x 128 query rows (8 waves x 16).  K/V 32-key tiles are
// TDM-loaded into double-buffered LDS by wave 0 and shared by all waves.
__global__ __launch_bounds__(256) void flash_attn(
    const unsigned short* __restrict__ qkv,
    unsigned short* __restrict__ O) {
  __shared__ unsigned short Ktile[2][32 * 64];   // [key][dh]
  __shared__ unsigned short Vtile[2][64 * 32];   // [dh][key]   (V^T tile)
  __shared__ unsigned short Ptile[8][16 * 32];   // per-wave P relayout

  const int lane = threadIdx.x & 31;
  const int wave = threadIdx.x >> 5;
  const int b  = blockIdx.x >> 8;
  const int h  = (blockIdx.x >> 4) & (NH - 1);
  const int qg = blockIdx.x & 15;
  const int qbase = qg * 128 + wave * 16;

  const size_t plane = (size_t)Bb * NH * Tq * DH;
  const size_t headoff = ((size_t)(b * NH + h)) * Tq * DH;
  const unsigned short* Q  = qkv + headoff;
  const unsigned short* K  = qkv + plane + headoff;
  const unsigned short* Vt = qkv + 2 * plane + headoff;  // [DH][Tq]

  const int rA = lane & 15, hA = lane >> 4;

  // Q tile (16 x 64) as two A fragments
  Frag16 qf[2];
#pragma unroll
  for (int ki = 0; ki < 2; ++ki) {
    const unsigned short* qp = Q + (size_t)(qbase + rA) * DH + 32 * ki;
#pragma unroll
    for (int j = 0; j < 8; ++j)
      qf[ki].u[j] = *(const unsigned*)(qp + a_koff(j, hA));
  }

  FragC acc[4];
#pragma unroll
  for (int f = 0; f < 4; ++f)
#pragma unroll
    for (int r = 0; r < 8; ++r) acc[f].f[r] = 0.0f;
  float mrow[8], lrow[8];
#pragma unroll
  for (int r = 0; r < 8; ++r) { mrow[r] = NEG_BIG; lrow[r] = 0.0f; }

  const int ntiles_blk = qg * 4 + 4;              // block causal extent
  const int ntiles_w   = (qbase + 47) >> 5;       // this wave's extent

  if (wave == 0) {                                // prime tile 0
    tdm_load_2d((unsigned)(size_t)(void*)&Ktile[0][0], K, DH, 32, DH);
    tdm_load_2d((unsigned)(size_t)(void*)&Vtile[0][0], Vt, 32, DH, Tq);
  }

  for (int jt = 0; jt < ntiles_blk; ++jt) {
    const int kb = jt * 32;
    if (wave == 0) {
      if (jt + 1 < ntiles_blk) {                  // prefetch next tile
        const int nb = (jt + 1) & 1, kn = kb + 32;
        tdm_load_2d((unsigned)(size_t)(void*)&Ktile[nb][0],
                    K + (size_t)kn * DH, DH, 32, DH);
        tdm_load_2d((unsigned)(size_t)(void*)&Vtile[nb][0],
                    Vt + kn, 32, DH, Tq);
        __builtin_amdgcn_s_wait_tensorcnt(2);     // tile jt landed
      } else {
        __builtin_amdgcn_s_wait_tensorcnt(0);
      }
    }
    __syncthreads();                              // tile jt visible to all

    if (jt < ntiles_w) {
      const unsigned short* Kl = &Ktile[jt & 1][0];
      const unsigned short* Vl = &Vtile[jt & 1][0];

      // S = Q * K^T  (16 x 32)
      FragC S[2];
#pragma unroll
      for (int s = 0; s < 2; ++s) {
#pragma unroll
        for (int r = 0; r < 8; ++r) S[s].f[r] = 0.0f;
#pragma unroll
        for (int ki = 0; ki < 2; ++ki) {
          Frag16 kf;  // B col n = key row (contiguous in LDS K tile)
          const unsigned short* kp =
              Kl + (16 * s + rA) * DH + 32 * ki + 16 * hA;
#pragma unroll
          for (int j = 0; j < 8; ++j)
            kf.u[j] = *(const unsigned*)(kp + 2 * j);
          S[s].v = wmma_bf16(qf[ki], kf, S[s].v);
        }
      }

      // causal mask + online softmax
      const int ncol = lane & 15;
#pragma unroll
      for (int r = 0; r < 8; ++r) {
        const int qi = qbase + r + 8 * hA;
#pragma unroll
        for (int s = 0; s < 2; ++s) {
          int kj = kb + 16 * s + ncol;
          if (kj > qi) S[s].f[r] = NEG_BIG;
        }
        float mx = fmaxf(S[0].f[r], S[1].f[r]);
#pragma unroll
        for (int d = 1; d < 16; d <<= 1)
          mx = fmaxf(mx, __shfl_xor(mx, d, 32));
        float mnew = fmaxf(mrow[r], mx);
        float alpha = __expf(mrow[r] - mnew);
        mrow[r] = mnew;
        float rsum = 0.0f;
#pragma unroll
        for (int s = 0; s < 2; ++s) {
          float p = __expf(S[s].f[r] - mnew);
          S[s].f[r] = p;
          rsum += p;
        }
#pragma unroll
        for (int d = 1; d < 16; d <<= 1)
          rsum += __shfl_xor(rsum, d, 32);
        lrow[r] = lrow[r] * alpha + rsum;
#pragma unroll
        for (int f = 0; f < 4; ++f) acc[f].f[r] *= alpha;
      }

      // P: C-fragment -> LDS -> A-fragment (intra-wave, DS in-order)
      unsigned short* pl = Ptile[wave];
#pragma unroll
      for (int r = 0; r < 8; ++r)
#pragma unroll
        for (int s = 0; s < 2; ++s)
          pl[(r + 8 * hA) * 32 + 16 * s + ncol] = f2bf(S[s].f[r]);
      __builtin_amdgcn_wave_barrier();
      Frag16 pf;
#pragma unroll
      for (int j = 0; j < 8; ++j)
        pf.u[j] = *(const unsigned*)(pl + rA * 32 + a_koff(j, hA));

      // O += P(16x32) * V(32x64); V^T tile makes B columns contiguous
#pragma unroll
      for (int f = 0; f < 4; ++f) {
        Frag16 vf;
        const unsigned short* vp = Vl + (16 * f + rA) * 32 + 16 * hA;
#pragma unroll
        for (int j = 0; j < 8; ++j)
          vf.u[j] = *(const unsigned*)(vp + 2 * j);
        acc[f].v = wmma_bf16(pf, vf, acc[f].v);
      }
    }
    __syncthreads();                              // done reading tile jt
  }

  // normalize, store O as bf16 in [B][T][DM]
#pragma unroll
  for (int r = 0; r < 8; ++r) {
    float inv = 1.0f / lrow[r];
    int m = qbase + r + 8 * hA;
#pragma unroll
    for (int f = 0; f < 4; ++f) {
      int c = h * DH + 16 * f + (lane & 15);
      O[((size_t)(b * Tq + m)) * DM + c] = f2bf(acc[f].f[r] * inv);
    }
  }
}

// ---------------- kernel 4: output projection GEMM ----------
__global__ __launch_bounds__(256) void out_gemm(
    const unsigned short* __restrict__ ob,
    const unsigned short* __restrict__ wb,
    float* __restrict__ out) {
  const int lane = threadIdx.x & 31;
  const int wave = threadIdx.x >> 5;
  const int wm = wave >> 2, wn = wave & 3;
  const int row0 = blockIdx.x * 64 + wm * 32;
  const int col0 = blockIdx.y * 128 + wn * 32;
  const int rA = lane & 15, hA = lane >> 4;

  FragC acc[2][2];
#pragma unroll
  for (int i = 0; i < 2; ++i)
#pragma unroll
    for (int j = 0; j < 2; ++j)
#pragma unroll
      for (int r = 0; r < 8; ++r) acc[i][j].f[r] = 0.0f;

  for (int k = 0; k < DM; k += 32) {
    Frag16 a[2], b[2];
#pragma unroll
    for (int i = 0; i < 2; ++i) {
      const unsigned short* ap = ob + (size_t)(row0 + 16 * i + rA) * DM + k;
      __builtin_prefetch(ap + 32, 0, 1);
#pragma unroll
      for (int j = 0; j < 8; ++j)
        a[i].u[j] = *(const unsigned*)(ap + a_koff(j, hA));
    }
#pragma unroll
    for (int i = 0; i < 2; ++i) {
      const unsigned short* bp =
          wb + (size_t)(col0 + 16 * i + rA) * DM + k + 16 * hA;
      __builtin_prefetch(bp + 32, 0, 1);
#pragma unroll
      for (int j = 0; j < 8; ++j)
        b[i].u[j] = *(const unsigned*)(bp + 2 * j);
    }
#pragma unroll
    for (int i = 0; i < 2; ++i)
#pragma unroll
      for (int j = 0; j < 2; ++j)
        acc[i][j].v = wmma_bf16(a[i], b[j], acc[i][j].v);
  }

#pragma unroll
  for (int i = 0; i < 2; ++i)
#pragma unroll
    for (int j = 0; j < 2; ++j)
#pragma unroll
      for (int r = 0; r < 8; ++r) {
        int m = row0 + 16 * i + r + 8 * hA;
        int n = col0 + 16 * j + rA;
        out[(size_t)m * DM + n] = acc[i][j].f[r];
      }
}

// ---------------- host launcher ----------------
extern "C" void kernel_launch(void* const* d_in, const int* in_sizes, int n_in,
                              void* d_out, int out_size, void* d_ws,
                              size_t ws_size, hipStream_t stream) {
  const float* x     = (const float*)d_in[0];
  const float* w_qkv = (const float*)d_in[1];
  const float* w_out = (const float*)d_in[2];
  float* out = (float*)d_out;

  const size_t nX  = (size_t)Bb * Tq * DM;
  const size_t nWq = (size_t)3 * DM * DM;
  const size_t nWo = (size_t)DM * DM;

  unsigned short* xb    = (unsigned short*)d_ws;
  unsigned short* wqkvb = xb + nX;
  unsigned short* woutb = wqkvb + nWq;
  unsigned short* qkvb  = woutb + nWo;
  unsigned short* Ob    = qkvb + 3 * nX;

  cast_bf16<<<(int)((nX + 255) / 256), 256, 0, stream>>>(x, xb, (int)nX);
  cast_bf16<<<(int)((nWq + 255) / 256), 256, 0, stream>>>(w_qkv, wqkvb, (int)nWq);
  cast_bf16<<<(int)((nWo + 255) / 256), 256, 0, stream>>>(w_out, woutb, (int)nWo);

  qkv_gemm<<<dim3(128, 24), 256, 0, stream>>>(xb, wqkvb, qkvb);
  flash_attn<<<1024, 256, 0, stream>>>(qkvb, Ob);
  out_gemm<<<dim3(128, 8), 256, 0, stream>>>(Ob, woutb, out);
}
